// MultiHeadedAttention_89300960018627
// MI455X (gfx1250) — compile-verified, matrix operand order checked by reading, not run on hardware
//
#include <hip/hip_runtime.h>
#include <math.h>

// Problem constants (from reference): B=16, T=512, F=1024, H=16, DK=64
#define B_  16
#define T_  512
#define F_  1024
#define H_  16
#define DK_ 64

typedef __attribute__((ext_vector_type(16))) __bf16 v16bf;
typedef __attribute__((ext_vector_type(8)))  float  v8f;
typedef __attribute__((ext_vector_type(4)))  unsigned int u32x4;
typedef __attribute__((ext_vector_type(8)))  int i32x8;
typedef __attribute__((ext_vector_type(4)))  int i32x4;

__device__ __forceinline__ v8f wmma_bf16(v16bf a, v16bf b, v8f c) {
  return __builtin_amdgcn_wmma_f32_16x16x32_bf16(false, a, false, b, (short)0, c,
                                                 false, false);
}

// ---- fragment loaders (ISA 7.12.2 layouts, wave32) -------------------------

__device__ __forceinline__ v16bf load_frag_a_f32(const float* base, int lda) {
  int lane = threadIdx.x & 31;
  int row  = lane & 15;
  int kh   = (lane >> 4) * 8;
  const float* p = base + (size_t)row * lda + kh;
  v16bf f;
#pragma unroll
  for (int v = 0; v < 8; ++v) {
    int k = (v & 4) ? (16 + (v & 3) * 2) : (v * 2);
    float2 d = *(const float2*)(p + k);
    f[2 * v]     = (__bf16)d.x;
    f[2 * v + 1] = (__bf16)d.y;
  }
  return f;
}

__device__ __forceinline__ v16bf load_frag_a_bf16(const __bf16* base, int lda) {
  int lane = threadIdx.x & 31;
  int row  = lane & 15;
  int kh   = (lane >> 4) * 8;
  const __bf16* p = base + (size_t)row * lda + kh;
  v16bf f;
#pragma unroll
  for (int v = 0; v < 8; ++v) {
    int k = (v & 4) ? (16 + (v & 3) * 2) : (v * 2);
    f[2 * v]     = p[k];
    f[2 * v + 1] = p[k + 1];
  }
  return f;
}

// B fragment from a K-major (KxN row-major) bf16 source (e.g. v[s][d] in LDS)
__device__ __forceinline__ v16bf load_frag_b_kxn(const __bf16* base, int ldb) {
  int lane = threadIdx.x & 31;
  int col  = lane & 15;
  int kh   = (lane >> 4) * 8;
  v16bf f;
#pragma unroll
  for (int j = 0; j < 16; ++j) {
    int k = ((j & 8) << 1) + (((j >> 1) & 3) * 2) + (j & 1) + kh;
    f[j] = base[(size_t)k * ldb + col];
  }
  return f;
}

__device__ __forceinline__ v8f load_c_f32(const float* base, int ldc) {
  int lane = threadIdx.x & 31;
  int col  = lane & 15;
  int rh   = (lane >> 4) * 8;
  v8f c;
#pragma unroll
  for (int r = 0; r < 8; ++r) c[r] = base[(size_t)(rh + r) * ldc + col];
  return c;
}

__device__ __forceinline__ void store_c_f32(float* base, int ldc, v8f c) {
  int lane = threadIdx.x & 31;
  int col  = lane & 15;
  int rh   = (lane >> 4) * 8;
#pragma unroll
  for (int r = 0; r < 8; ++r) base[(size_t)(rh + r) * ldc + col] = c[r];
}

// ---- Tensor Data Mover staging (ISA ch.8 async tensor) ---------------------
// Load a 2-D bf16 tile (tile_x cols x tile_y rows, row stride = stride_x
// elements) from global into LDS. D# packed per ISA 8.3/8.4:
//   group0: count=1 | lds_addr | global_addr | type=2
//   group1: data_size=2B, tensor_dim0/1, tile_dim0/1, dim strides
// This toolchain declares the 6-arg builtin (extra int32x8 descriptor group).
__device__ __forceinline__ void tdm_load_tile_bf16(unsigned lds_off,
                                                   const __bf16* gptr,
                                                   unsigned tile_x, unsigned tile_y,
                                                   unsigned stride_x) {
#if __has_builtin(__builtin_amdgcn_tensor_load_to_lds) && __has_builtin(__builtin_amdgcn_s_wait_tensorcnt)
  unsigned long long ga = (unsigned long long)gptr;
  unsigned td0 = stride_x;
  unsigned td1 = 0x40000000u;                       // huge dim1: no OOB clipping
  unsigned long long s0 = stride_x;                 // tensor_dim0_stride
  unsigned long long s1 = (unsigned long long)stride_x * tile_y;
  u32x4 g0;
  g0.x = 1u;                                        // count=1, user mode
  g0.y = lds_off;                                   // lds_addr (bytes)
  g0.z = (unsigned)(ga & 0xffffffffu);              // global_addr[31:0]
  g0.w = (unsigned)((ga >> 32) & 0x1ffffffu) | (2u << 30);  // addr[56:32]|type=2
  i32x8 g1;
  g1[0] = (int)(1u << 16);                                  // data_size=2B
  g1[1] = (int)((td0 & 0xffffu) << 16);                     // tensor_dim0 lo
  g1[2] = (int)((td0 >> 16) | ((td1 & 0xffffu) << 16));     // td0 hi | td1 lo
  g1[3] = (int)((td1 >> 16) | (tile_x << 16));              // td1 hi | tile_dim0
  g1[4] = (int)(tile_y & 0xffffu);                          // tile_dim1, tile_dim2=0
  g1[5] = (int)(unsigned)(s0 & 0xffffffffu);                // dim0 stride lo
  g1[6] = (int)((unsigned)(s0 >> 32) | ((unsigned)(s1 & 0xffffu) << 16));
  g1[7] = (int)(unsigned)((s1 >> 16) & 0xffffffffu);        // dim1 stride hi
  i32x4 z4 = {0, 0, 0, 0};
  i32x8 z8 = {0, 0, 0, 0, 0, 0, 0, 0};
  __builtin_amdgcn_tensor_load_to_lds(g0, g1, z4, z4, z8, 0);
  __builtin_amdgcn_s_wait_tensorcnt(0);
#endif
}

// block-cooperative stage of a contiguous n-element bf16 slab into LDS
__device__ __forceinline__ void stage_slab(__bf16* lds, const __bf16* g, int n) {
#if __has_builtin(__builtin_amdgcn_tensor_load_to_lds) && __has_builtin(__builtin_amdgcn_s_wait_tensorcnt)
  if (threadIdx.x < 32) {   // one wave issues the TDM DMA and waits TENSORcnt
    tdm_load_tile_bf16((unsigned)(unsigned long long)(void*)lds, g, DK_,
                       (unsigned)(n / DK_), DK_);
  }
#else
  for (int i = threadIdx.x; i < n; i += blockDim.x) lds[i] = g[i];
#endif
  __syncthreads();
}

// ---- kernels ---------------------------------------------------------------

__global__ void mha_cvt_bf16(const float* __restrict__ in, __bf16* __restrict__ out,
                             size_t n) {
  size_t i = (size_t)blockIdx.x * blockDim.x + threadIdx.x;
  size_t stride = (size_t)gridDim.x * blockDim.x;
  for (; i < n; i += stride) out[i] = (__bf16)in[i];
}

__device__ __forceinline__ void store_proj_tile(__bf16* out, const float* bias,
                                                int m0, int f0, v8f c) {
  int lane = threadIdx.x & 31;
  int col = lane & 15, rh = (lane >> 4) * 8;
  int f = f0 + col;
  float bv = bias[f];
  int h = f >> 6, d = f & 63;
#pragma unroll
  for (int r = 0; r < 8; ++r) {
    int bt = m0 + rh + r;
    int b_ = bt >> 9, t = bt & 511;
    out[(((size_t)b_ * H_ + h) * T_ + t) * DK_ + d] = (__bf16)(c[r] + bv);
  }
}

// out[b][h][t][d] (bf16) = X @ W^T + bias.  M=8192, N=1024, K=1024.
// 2x2 register blocking: one wave computes a 32x32 C block (4 WMMA / 4 frags).
__global__ void __launch_bounds__(256) mha_proj(const float* __restrict__ X,
                                                const float* __restrict__ W,
                                                const float* __restrict__ bias,
                                                __bf16* __restrict__ out) {
  int gw = (blockIdx.x * blockDim.x + threadIdx.x) >> 5;
  int nt = gw & 31;                 // 32 n-blocks of 32
  int mt = gw >> 5;                 // 256 m-blocks of 32
  int m0 = mt * 32, f0 = nt * 32;
  v8f c00 = {}, c01 = {}, c10 = {}, c11 = {};
  for (int kk = 0; kk < F_; kk += 32) {
    __builtin_prefetch(X + (size_t)m0 * F_ + kk + 256, 0, 0);
    v16bf a0 = load_frag_a_f32(X + (size_t)m0 * F_ + kk, F_);
    v16bf a1 = load_frag_a_f32(X + (size_t)(m0 + 16) * F_ + kk, F_);
    v16bf b0 = load_frag_a_f32(W + (size_t)f0 * F_ + kk, F_);
    v16bf b1 = load_frag_a_f32(W + (size_t)(f0 + 16) * F_ + kk, F_);
    c00 = wmma_bf16(a0, b0, c00);
    c01 = wmma_bf16(a0, b1, c01);
    c10 = wmma_bf16(a1, b0, c10);
    c11 = wmma_bf16(a1, b1, c11);
  }
  store_proj_tile(out, bias, m0, f0, c00);
  store_proj_tile(out, bias, m0, f0 + 16, c01);
  store_proj_tile(out, bias, m0 + 16, f0, c10);
  store_proj_tile(out, bias, m0 + 16, f0 + 16, c11);
}

// S[bh][t][s] = q[bh] @ k[bh]^T : per-(b,h) 512x512x64, 2x2 blocking.
__global__ void __launch_bounds__(256) mha_score_content(const __bf16* __restrict__ qb,
                                                         const __bf16* __restrict__ kb,
                                                         float* __restrict__ S) {
  int gw  = (blockIdx.x * blockDim.x + threadIdx.x) >> 5;
  int st2 = gw & 15;
  int tt2 = (gw >> 4) & 15;
  int bh  = gw >> 8;
  const __bf16* Aq = qb + ((size_t)bh * T_ + tt2 * 32) * DK_;
  const __bf16* Bk = kb + ((size_t)bh * T_ + st2 * 32) * DK_;
  v8f c00 = {}, c01 = {}, c10 = {}, c11 = {};
#pragma unroll
  for (int kk = 0; kk < DK_; kk += 32) {
    v16bf a0 = load_frag_a_bf16(Aq + kk, DK_);
    v16bf a1 = load_frag_a_bf16(Aq + (size_t)16 * DK_ + kk, DK_);
    v16bf b0 = load_frag_a_bf16(Bk + kk, DK_);
    v16bf b1 = load_frag_a_bf16(Bk + (size_t)16 * DK_ + kk, DK_);
    c00 = wmma_bf16(a0, b0, c00);
    c01 = wmma_bf16(a0, b1, c01);
    c10 = wmma_bf16(a1, b0, c10);
    c11 = wmma_bf16(a1, b1, c11);
  }
  float* Cp = S + (size_t)bh * T_ * T_ + (size_t)(tt2 * 32) * T_ + st2 * 32;
  store_c_f32(Cp, T_, c00);
  store_c_f32(Cp + 16, T_, c01);
  store_c_f32(Cp + (size_t)16 * T_, T_, c10);
  store_c_f32(Cp + (size_t)16 * T_ + 16, T_, c11);
}

// S[bh][t][s] += Q_t @ pos_k[t]^T.  Block = (t, half of bh-tiles); the 64KB
// pos_k[t] slab is TDM-DMA'd into LDS once and shared by all 8 waves. Each
// wave loads its A fragments once and sweeps all 32 s-tiles from LDS.
__global__ void __launch_bounds__(256) mha_score_rel(const __bf16* __restrict__ qb,
                                                     const __bf16* __restrict__ pkb,
                                                     float* __restrict__ S) {
  __shared__ __bf16 Btile[T_ * DK_];  // 64 KB: pos_k[t] (512 x 64 bf16)
  int t     = blockIdx.x >> 1;
  int bhgrp = blockIdx.x & 1;
  int wave  = threadIdx.x >> 5;
  stage_slab(Btile, pkb + (size_t)t * T_ * DK_, T_ * DK_);

  int bht = bhgrp * 8 + wave;                       // 16 bh rows per wave-tile
  const __bf16* A = qb + ((size_t)(bht * 16) * T_ + t) * DK_;   // lda = T_*DK_
  v16bf a0 = load_frag_a_bf16(A, T_ * DK_);
  v16bf a1 = load_frag_a_bf16(A + 32, T_ * DK_);
  for (int st = 0; st < 32; ++st) {
    const __bf16* Bp = Btile + (size_t)(st * 16) * DK_;         // LDS (ds_load)
    float* Cp = S + (size_t)(bht * 16) * T_ * T_ + (size_t)t * T_ + st * 16;
    v8f c = load_c_f32(Cp, T_ * T_);
    c = wmma_bf16(a0, load_frag_a_bf16(Bp, DK_), c);
    c = wmma_bf16(a1, load_frag_a_bf16(Bp + 32, DK_), c);
    store_c_f32(Cp, T_ * T_, c);
  }
}

// masked softmax over s, one wave per (bh,t) row; writes attn as bf16
__global__ void __launch_bounds__(256) mha_softmax(const float* __restrict__ S,
                                                   const int* __restrict__ mask,
                                                   __bf16* __restrict__ P) {
  int wid  = (blockIdx.x * blockDim.x + threadIdx.x) >> 5;
  int lane = threadIdx.x & 31;
  int t  = wid & 511;
  int bh = wid >> 9;
  int b  = bh >> 4;
  const float* Sr = S + (size_t)bh * T_ * T_ + (size_t)t * T_;
  const int*   Mr = mask + ((size_t)b * T_ + t) * T_;
  float vals[16];
  float mx = -3.402823466e38f;
#pragma unroll
  for (int i = 0; i < 16; ++i) {
    int s = lane + 32 * i;
    float v = Sr[s] * 0.125f;                      // 1/sqrt(DK)
    if (Mr[s] == 0) v = -3.402823466e38f;
    vals[i] = v;
    mx = fmaxf(mx, v);
  }
  for (int off = 16; off > 0; off >>= 1) mx = fmaxf(mx, __shfl_xor(mx, off, 32));
  float sum = 0.f;
#pragma unroll
  for (int i = 0; i < 16; ++i) {
    float e = __expf(vals[i] - mx);
    vals[i] = e;
    sum += e;
  }
  for (int off = 16; off > 0; off >>= 1) sum += __shfl_xor(sum, off, 32);
  float inv = 1.f / sum;
  __bf16* Pr = P + (size_t)bh * T_ * T_ + (size_t)t * T_;
#pragma unroll
  for (int i = 0; i < 16; ++i) {
    int s = lane + 32 * i;
    float a = (Mr[s] == 0) ? 0.f : vals[i] * inv;
    Pr[s] = (__bf16)a;
  }
}

// xb[b][t][h*DK+d] = P[bh] @ v[bh].  Block = (bh, half of t); v[bh] slab
// (64KB) TDM'd into LDS.  Each wave: 2 t-tiles x 4 d-tiles, A reused 4x,
// B fragments reused 2x.
__global__ void __launch_bounds__(256) mha_av_content(const __bf16* __restrict__ P,
                                                      const __bf16* __restrict__ vb,
                                                      float* __restrict__ xb) {
  __shared__ __bf16 Btile[T_ * DK_];  // 64 KB: v[bh] (512 x 64 bf16)
  int bh   = blockIdx.x >> 1;
  int half = blockIdx.x & 1;
  int wave = threadIdx.x >> 5;
  stage_slab(Btile, vb + (size_t)bh * T_ * DK_, T_ * DK_);

  int b_ = bh >> 4, h = bh & 15;
  int tt0 = half * 16 + wave * 2;
  const __bf16* A0 = P + (size_t)bh * T_ * T_ + (size_t)(tt0 * 16) * T_;
  const __bf16* A1 = A0 + (size_t)16 * T_;
  v8f c[2][4];
#pragma unroll
  for (int i = 0; i < 2; ++i)
#pragma unroll
    for (int dt = 0; dt < 4; ++dt) c[i][dt] = (v8f){};
  for (int kk = 0; kk < T_; kk += 32) {
    v16bf a0 = load_frag_a_bf16(A0 + kk, T_);
    v16bf a1 = load_frag_a_bf16(A1 + kk, T_);
#pragma unroll
    for (int dt = 0; dt < 4; ++dt) {
      v16bf bb = load_frag_b_kxn(Btile + (size_t)kk * DK_ + dt * 16, DK_);
      c[0][dt] = wmma_bf16(a0, bb, c[0][dt]);
      c[1][dt] = wmma_bf16(a1, bb, c[1][dt]);
    }
  }
#pragma unroll
  for (int i = 0; i < 2; ++i) {
    float* Cp = xb + ((size_t)b_ * T_ + (tt0 + i) * 16) * F_ + h * DK_;
#pragma unroll
    for (int dt = 0; dt < 4; ++dt) store_c_f32(Cp + dt * 16, F_, c[i][dt]);
  }
}

// xb += P_t @ pos_v[t].  Block = (t, half of bh-tiles); pos_v[t] slab TDM'd
// into LDS.  A fragment reused across the 4 d-tiles.
__global__ void __launch_bounds__(256) mha_av_rel(const __bf16* __restrict__ P,
                                                  const __bf16* __restrict__ pvb,
                                                  float* __restrict__ xb) {
  __shared__ __bf16 Btile[T_ * DK_];  // 64 KB: pos_v[t] (512 x 64 bf16)
  int t     = blockIdx.x >> 1;
  int bhgrp = blockIdx.x & 1;
  int wave  = threadIdx.x >> 5;
  stage_slab(Btile, pvb + (size_t)t * T_ * DK_, T_ * DK_);

  int bht = bhgrp * 8 + wave;                       // == b (H == 16)
  const __bf16* A = P + ((size_t)(bht * 16) * T_ + t) * T_;  // lda = T_*T_
  float* Cp = xb + ((size_t)bht * T_ + t) * F_;              // ldc = DK_ (over h)
  v8f c[4];
#pragma unroll
  for (int dt = 0; dt < 4; ++dt) c[dt] = load_c_f32(Cp + dt * 16, DK_);
  for (int kk = 0; kk < T_; kk += 32) {
    v16bf a = load_frag_a_bf16(A + kk, T_ * T_);
#pragma unroll
    for (int dt = 0; dt < 4; ++dt) {
      v16bf bb = load_frag_b_kxn(Btile + (size_t)kk * DK_ + dt * 16, DK_);
      c[dt] = wmma_bf16(a, bb, c[dt]);
    }
  }
#pragma unroll
  for (int dt = 0; dt < 4; ++dt) store_c_f32(Cp + dt * 16, DK_, c[dt]);
}

// out = xb @ Wo^T + bo (fp32 output), 2x2 register blocking
__global__ void __launch_bounds__(256) mha_out_proj(const float* __restrict__ xb,
                                                    const float* __restrict__ Wo,
                                                    const float* __restrict__ bo,
                                                    float* __restrict__ out) {
  int gw   = (blockIdx.x * blockDim.x + threadIdx.x) >> 5;
  int lane = threadIdx.x & 31;
  int nt = gw & 31;
  int mt = gw >> 5;
  int m0 = mt * 32, f0 = nt * 32;
  v8f c00 = {}, c01 = {}, c10 = {}, c11 = {};
  for (int kk = 0; kk < F_; kk += 32) {
    __builtin_prefetch(xb + (size_t)m0 * F_ + kk + 256, 0, 0);
    v16bf a0 = load_frag_a_f32(xb + (size_t)m0 * F_ + kk, F_);
    v16bf a1 = load_frag_a_f32(xb + (size_t)(m0 + 16) * F_ + kk, F_);
    v16bf b0 = load_frag_a_f32(Wo + (size_t)f0 * F_ + kk, F_);
    v16bf b1 = load_frag_a_f32(Wo + (size_t)(f0 + 16) * F_ + kk, F_);
    c00 = wmma_bf16(a0, b0, c00);
    c01 = wmma_bf16(a0, b1, c01);
    c10 = wmma_bf16(a1, b0, c10);
    c11 = wmma_bf16(a1, b1, c11);
  }
  int col = lane & 15, rh = (lane >> 4) * 8;
#pragma unroll
  for (int r = 0; r < 8; ++r) {
    out[(size_t)(m0 + rh + r) * F_ + f0 + col]           = c00[r] + bo[f0 + col];
    out[(size_t)(m0 + rh + r) * F_ + f0 + 16 + col]      = c01[r] + bo[f0 + 16 + col];
    out[(size_t)(m0 + 16 + rh + r) * F_ + f0 + col]      = c10[r] + bo[f0 + col];
    out[(size_t)(m0 + 16 + rh + r) * F_ + f0 + 16 + col] = c11[r] + bo[f0 + 16 + col];
  }
}

// ---- host launch -----------------------------------------------------------
extern "C" void kernel_launch(void* const* d_in, const int* in_sizes, int n_in,
                              void* d_out, int out_size, void* d_ws, size_t ws_size,
                              hipStream_t stream) {
  const float* query = (const float*)d_in[0];
  const float* key_  = (const float*)d_in[1];
  const float* value = (const float*)d_in[2];
  const float* pos_k = (const float*)d_in[3];
  const float* pos_v = (const float*)d_in[4];
  const int*   mask  = (const int*)d_in[5];
  const float* Wq = (const float*)d_in[6];
  const float* bq = (const float*)d_in[7];
  const float* Wk = (const float*)d_in[8];
  const float* bk = (const float*)d_in[9];
  const float* Wv = (const float*)d_in[10];
  const float* bv = (const float*)d_in[11];
  const float* Wo = (const float*)d_in[12];
  const float* bo = (const float*)d_in[13];
  float* out = (float*)d_out;

  char* ws = (char*)d_ws;
  size_t off = 0;
  auto carve = [&](size_t bytes) {
    void* p = ws + off;
    off += (bytes + 255) & ~(size_t)255;
    return p;
  };
  const size_t qkv_b = (size_t)B_ * H_ * T_ * DK_ * sizeof(__bf16);  // 16 MiB
  const size_t pos_b = (size_t)T_ * T_ * DK_ * sizeof(__bf16);       // 32 MiB
  const size_t S_b   = (size_t)B_ * H_ * T_ * T_ * sizeof(float);    // 256 MiB
  const size_t P_b   = (size_t)B_ * H_ * T_ * T_ * sizeof(__bf16);   // 128 MiB
  const size_t xb_b  = (size_t)B_ * T_ * F_ * sizeof(float);         // 32 MiB

  __bf16* qb   = (__bf16*)carve(qkv_b);
  __bf16* kb   = (__bf16*)carve(qkv_b);
  __bf16* vbuf = (__bf16*)carve(qkv_b);
  __bf16* pkb  = (__bf16*)carve(pos_b);
  __bf16* pvb  = (__bf16*)carve(pos_b);
  float*  S    = (float*)carve(S_b);
  __bf16* P    = (__bf16*)carve(P_b);
  float*  xb   = (float*)carve(xb_b);

  dim3 blk(256);  // 8 waves (wave32)

  mha_cvt_bf16<<<4096, blk, 0, stream>>>(pos_k, pkb, (size_t)T_ * T_ * DK_);
  mha_cvt_bf16<<<4096, blk, 0, stream>>>(pos_v, pvb, (size_t)T_ * T_ * DK_);

  // projections: 8192 wave-blocks (32x32 tiles) -> 1024 blocks
  mha_proj<<<1024, blk, 0, stream>>>(query, Wq, bq, qb);
  mha_proj<<<1024, blk, 0, stream>>>(key_,  Wk, bk, kb);
  mha_proj<<<1024, blk, 0, stream>>>(value, Wv, bv, vbuf);

  // content scores: 65536 wave-blocks -> 8192 blocks
  mha_score_content<<<8192, blk, 0, stream>>>(qb, kb, S);
  // relative scores: block per (t, bh-half), TDM-staged pos_k[t]
  mha_score_rel<<<1024, blk, 0, stream>>>(qb, pkb, S);

  // softmax: 131072 rows, one wave each -> 16384 blocks
  mha_softmax<<<16384, blk, 0, stream>>>(S, mask, P);

  // attention x value: block per (bh, t-half), TDM-staged v[bh]
  mha_av_content<<<512, blk, 0, stream>>>(P, vbuf, xb);
  // relative value: block per (t, bh-half), TDM-staged pos_v[t]
  mha_av_rel<<<1024, blk, 0, stream>>>(P, pvb, xb);

  // output projection -> d_out (fp32)
  mha_out_proj<<<1024, blk, 0, stream>>>(xb, Wo, bo, out);
}
